// KMeansQuantizer_85890755985616
// MI455X (gfx1250) — matrix operand order repaired and checked by reading, not compiled, not run on hardware
//
#include <hip/hip_runtime.h>
#include <hip/hip_bf16.h>

typedef __attribute__((ext_vector_type(16))) __bf16 v16bf;
typedef __attribute__((ext_vector_type(4)))  __bf16 v4bf;
typedef __attribute__((ext_vector_type(8)))  float  v8f;
typedef __attribute__((ext_vector_type(4)))  float  v4f;
typedef __attribute__((ext_vector_type(4)))  unsigned int v4u;

#define DIM   768
#define KCLUS 500
#define KPAD  512   // padded center count (multiple of 64)
#define ROWS_PER_BLOCK 32   // two 16-row M-tiles per block: B fragments reused 2x

// 32-byte container whose register image is exactly a WMMA 16xbf16 operand
// (low 4 VGPRs = first 16B load, high 4 VGPRs = second 16B load).
struct Frag32B { v4u lo; v4u hi; };

static __device__ __forceinline__ v16bf load_frag(const char* p0, const char* p1) {
    Frag32B f;
    f.lo = *(const v4u*)p0;
    f.hi = *(const v4u*)p1;
    return __builtin_bit_cast(v16bf, f);
}

// ---------------------------------------------------------------------------
// Kernel 1: convert centers f32 -> bf16 [KPAD][DIM] (pad rows zeroed) and
// compute ||c||^2 in f32 from the ORIGINAL f32 centers. Pad rows get +huge
// so they can never win the argmin.
// ---------------------------------------------------------------------------
__global__ __launch_bounds__(256) void prep_centers_kernel(
    const float* __restrict__ centers,  // [KCLUS][DIM]
    __bf16* __restrict__ cb,            // [KPAD][DIM]
    float* __restrict__ csq)            // [KPAD]
{
    const int k = blockIdx.x;   // 0..KPAD-1
    const int t = threadIdx.x;
    __shared__ float red[256];

    float s = 0.0f;
    if (k < KCLUS) {
        for (int d = t; d < DIM; d += 256) {
            float v = centers[k * DIM + d];
            cb[k * DIM + d] = (__bf16)v;
            s += v * v;
        }
    } else {
        for (int d = t; d < DIM; d += 256) {
            cb[k * DIM + d] = (__bf16)0.0f;
        }
    }
    red[t] = s;
    __syncthreads();
    for (int off = 128; off > 0; off >>= 1) {
        if (t < off) red[t] += red[t + off];
        __syncthreads();
    }
    if (t == 0) csq[k] = (k < KCLUS) ? red[0] : 3.0e38f;
}

// ---------------------------------------------------------------------------
// Kernel 2: fused GEMM (bf16 WMMA, f32 accum) + argmin.
// Block = 256 threads = 8 wave32. Block handles 32 rows x 512 centers
// (two 16-row M-tiles sharing every B fragment). Wave w owns center
// columns [w*64, w*64+64).
// ---------------------------------------------------------------------------
__global__ __launch_bounds__(256) void kmeans_assign_kernel(
    const float* __restrict__ x,        // [N][DIM], N = 65536
    const __bf16* __restrict__ cb,      // [KPAD][DIM]
    const float* __restrict__ csq,      // [KPAD]
    long long* __restrict__ out)        // [N]
{
    // A tile (48 KB, live during the K-loop) unioned with the per-pass
    // 16-row distance tile (32 KB, live only in the epilogue).
    __shared__ alignas(16) union {
        __bf16 a[ROWS_PER_BLOCK * DIM];   // bf16 x-tile
        float  d[16 * KPAD];              // one 16-row distance tile
    } sm;
    __shared__ float redV[256];
    __shared__ int   redI[256];

    const int t = threadIdx.x;
    const long long rowbase = (long long)blockIdx.x * ROWS_PER_BLOCK;

    // ---- Stage x tile into LDS as bf16. Non-temporal: x is streamed once,
    // keep it out of the L2 the centers live in. 16B loads + packed
    // 8-byte bf16 stores (4x fewer staging instructions). ----
    const float* __restrict__ xrow = x + rowbase * DIM;
    for (int idx = t * 4; idx < ROWS_PER_BLOCK * DIM; idx += 256 * 4) {
        v4f v = __builtin_nontemporal_load((const v4f*)&xrow[idx]);
        v4bf h;
        h[0] = (__bf16)v[0]; h[1] = (__bf16)v[1];
        h[2] = (__bf16)v[2]; h[3] = (__bf16)v[3];
        *(v4bf*)&sm.a[idx] = h;
    }
    __syncthreads();

    const int lane    = t & 31;
    const int wave    = t >> 5;
    const int m       = lane & 15;         // row within A tile / col within B tile
    const int halfk8  = (lane >> 4) << 3;  // A-layout K offset: 0 or 8 (elements)
    const int halfk16 = (lane >> 4) << 4;  // unused (B also uses halfk8): 0 or 16 (elements)
    const int nbase   = wave * 64;

    // Preload ||c||^2 for the 4 columns this lane contributes to.
    float cs[4];
#pragma unroll
    for (int j = 0; j < 4; ++j) cs[j] = csq[nbase + j * 16 + m];

    // ---- Precompute 32-bit byte offsets once; the K-loop then indexes as
    // base + (off + kb*2) so unrolled iterations fold into immediate offsets
    // (ds_load_b128 voff offset:imm / global_load_b128 v, voff, s[cb] offset).
    const char* aB  = (const char*)sm.a;
    const char* cbB = (const char*)cb;
    const unsigned a0off = (unsigned)(m * DIM + halfk8) * 2u;
    const unsigned a1off = a0off + (unsigned)(16 * DIM) * 2u;
    unsigned boff[4];
#pragma unroll
    for (int j = 0; j < 4; ++j)
        boff[j] = ((unsigned)(nbase + j * 16 + m) * DIM + (unsigned)halfk8) * 2u;

    v8f acc0[4] = {};   // M-tile 0: rows 0..15
    v8f acc1[4] = {};   // M-tile 1: rows 16..31

#pragma unroll 4
    for (int kb = 0; kb < DIM; kb += 32) {
        const unsigned kb2 = (unsigned)kb * 2u;   // byte offset along K

        // A fragments (16x32 bf16). Lane<16: M=lane, K={kb..+7, kb+16..+23};
        // lane>=16: K shifted by 8. Halves are contiguous 16B LDS reads.
        const v16bf a0 = load_frag(aB + a0off + kb2, aB + a0off + kb2 + 32);
        const v16bf a1 = load_frag(aB + a1off + kb2, aB + a1off + kb2 + 32);

        // B fragments (32x16 bf16). Lane<16: N=lane, K={kb..+7, kb+16..+23};
        // lane>=16: K shifted by 8. Two 16B global loads (L2-hot).
        // Each B fragment feeds BOTH M-tiles.
#pragma unroll
        for (int j = 0; j < 4; ++j) {
            const v16bf b = load_frag(cbB + boff[j] + kb2,
                                      cbB + boff[j] + kb2 + 32);

            acc0[j] = __builtin_amdgcn_wmma_f32_16x16x32_bf16(
                false, a0, false, b, (short)0, acc0[j], false, false);
            acc1[j] = __builtin_amdgcn_wmma_f32_16x16x32_bf16(
                false, a1, false, b, (short)0, acc1[j], false, false);
        }
    }

    __syncthreads();   // A tile dead from here; sm.d may overwrite it.

    // ---- Epilogue: two 16-row passes of (dist -> LDS -> fused argmin).
    // C layout: VGPR i, lanes 0-15 -> M=i, lanes 16-31 -> M=i+8; N = lane&15.
    const int mofs = (lane >> 4) << 3;   // 0 or 8
    const int row  = t >> 4;             // 0..15 (row within the pass tile)
    const int part = t & 15;             // 16 scanners per row

#pragma unroll
    for (int p = 0; p < 2; ++p) {
        v8f (&acc)[4] = (p == 0) ? acc0 : acc1;

#pragma unroll
        for (int j = 0; j < 4; ++j) {
            const int N = nbase + j * 16 + m;
#pragma unroll
            for (int i = 0; i < 8; ++i) {
                sm.d[(i + mofs) * KPAD + N] = cs[j] - 2.0f * acc[j][i];
            }
        }
        __syncthreads();

        // First-min-index semantics like jnp.argmin.
        {
            const float* dr = &sm.d[row * KPAD + part * 32];
            float best = dr[0];
            int   bi   = 0;
#pragma unroll 4
            for (int c = 1; c < 32; ++c) {
                float v = dr[c];
                if (v < best) { best = v; bi = c; }   // strict < keeps lowest idx
            }
            redV[t] = best;
            redI[t] = part * 32 + bi;
        }
        __syncthreads();

        if (part == 0) {
            float bv  = redV[t];
            int   bid = redI[t];
#pragma unroll
            for (int q = 1; q < 16; ++q) {
                float v  = redV[t + q];
                int   id = redI[t + q];
                if (v < bv || (v == bv && id < bid)) { bv = v; bid = id; }
            }
            out[rowbase + p * 16 + row] = (long long)bid;
        }
        __syncthreads();   // protect sm.d / redV before next pass reuses them
    }
}

// ---------------------------------------------------------------------------
// Launch wrapper
// ---------------------------------------------------------------------------
extern "C" void kernel_launch(void* const* d_in, const int* in_sizes, int n_in,
                              void* d_out, int out_size, void* d_ws, size_t ws_size,
                              hipStream_t stream) {
    const float* x       = (const float*)d_in[0];   // [16,4096,768] f32
    const float* centers = (const float*)d_in[1];   // [500,768] f32

    // Workspace layout: bf16 centers [KPAD][DIM], then f32 csq [KPAD].
    __bf16* cb  = (__bf16*)d_ws;
    float*  csq = (float*)((char*)d_ws + (size_t)KPAD * DIM * sizeof(__bf16));

    long long* out = (long long*)d_out;             // int64 [16,4096]

    const int n_rows = in_sizes[0] / DIM;           // 65536

    prep_centers_kernel<<<KPAD, 256, 0, stream>>>(centers, cb, csq);
    kmeans_assign_kernel<<<n_rows / ROWS_PER_BLOCK, 256, 0, stream>>>(x, cb, csq, out);
}
